// TransformerBlock_20409684591294
// MI455X (gfx1250) — matrix operand order repaired and checked by reading, not compiled, not run on hardware
//
#include <hip/hip_runtime.h>
#include <cstdint>
#include <cstddef>

#define DIM     1024
#define HEADS   16
#define DH      64
#define NEXP    8
#define DFF     4096
#define BATCH   2
#define SEQ     2048
#define TOK     (BATCH*SEQ)
#define QKVLD   (3*DIM)
#define LDSPAD  40   // 32 cols + 8 pad -> 20-bank row skew, conflict-free b128 reads

typedef __bf16 bf16;
typedef __attribute__((ext_vector_type(8)))  __bf16 v8bf;
typedef __attribute__((ext_vector_type(16))) __bf16 v16bf;
typedef __attribute__((ext_vector_type(8)))  float  v8f;

static __device__ __forceinline__ v8f vzero8f() {
  v8f z;
#pragma unroll
  for (int i = 0; i < 8; ++i) z[i] = 0.0f;
  return z;
}
static __device__ __forceinline__ v16bf vzero16bf() {
  v16bf z;
#pragma unroll
  for (int i = 0; i < 16; ++i) z[i] = (bf16)0.0f;
  return z;
}
static __device__ __forceinline__ v16bf cat8(v8bf lo, v8bf hi) {
  return __builtin_shufflevector(lo, hi, 0,1,2,3,4,5,6,7,8,9,10,11,12,13,14,15);
}

// A fragment (16x32, row-major src). Lane L: row = L%16,
// K = (L/16)*8 + {0..7} and (L/16)*8 + {16..23}.  (ISA 7.12.2, 16-bit A)
static __device__ __forceinline__ v16bf load_a_frag(const bf16* base, int stride) {
  const int lane = threadIdx.x & 31;
  const bf16* p = base + (size_t)(lane & 15) * (size_t)stride + ((lane >> 4) << 3);
  return cat8(*(const v8bf*)p, *(const v8bf*)(p + 16));
}
// B fragment (32x16) loaded from B^T row-major (rows = N, cols = K).
// Lane L: col N = L%16, K = (L/16)*8 + {0..7} and (L/16)*8 + {16..23}. (ISA 7.12.2/7.12.4)
static __device__ __forceinline__ v16bf load_b_frag(const bf16* baseT, int stride) {
  const int lane = threadIdx.x & 31;
  const bf16* p = baseT + (size_t)(lane & 15) * (size_t)stride + ((lane >> 4) << 3);
  return cat8(*(const v8bf*)p, *(const v8bf*)(p + 16));
}
static __device__ __forceinline__ v8f wmma_bf16(v16bf a, v16bf b, v8f c) {
  return __builtin_amdgcn_wmma_f32_16x16x32_bf16(false, a, false, b, (short)0, c,
                                                 false, false);
}

// CDNA5 async global->LDS copy (ASYNCcnt). 16B per lane, wave-level instruction.
// GV addressing: 64-bit global addr in VGPR pair, LDS byte address in VDST VGPR.
static __device__ __forceinline__ void async_copy_b128(const void* gptr, void* lptr) {
  const uint32_t lds = (uint32_t)(uintptr_t)lptr;   // low 32 bits = LDS offset
  asm volatile("global_load_async_to_lds_b128 %0, %1, off"
               :: "v"(lds), "v"(gptr) : "memory");
}
static __device__ __forceinline__ void wait_async_le4() {
  asm volatile("s_wait_asynccnt 0x4" ::: "memory");
}
static __device__ __forceinline__ void wait_async_zero() {
  asm volatile("s_wait_asynccnt 0x0" ::: "memory");
}

// ---------------------------------------------------------------------------
// Generic NT GEMM: out[m,n] = sum_k A[m,k]*Bt[n,k]  (both bf16, f32 accum)
// Block 256 thr = 8 waves; tile 128x128; wave tile 64x32 (4x2 WMMA tiles).
// CONV=0: double-buffered LDS staging via global_load_async_to_lds_b128.
// CONV=1: A is xbf (T x D), logical K = tap*D + i with seq-shift & zero pad
//         (direct global fragment loads; zero rows can't be async-copied).
// ---------------------------------------------------------------------------
template <int CONV>
__global__ __launch_bounds__(256) void gemm_nt_kernel(
    const bf16* __restrict__ A, int lda,
    const bf16* __restrict__ Bt, int ldb,
    int M, int N, int K,
    const float* __restrict__ bias,
    float* __restrict__ Cf, bf16* __restrict__ Cb, int ldc,
    int relu,
    const float* __restrict__ rowscale, int rs_stride,
    float* __restrict__ Cacc)
{
  const int tid  = threadIdx.x;
  const int lane = tid & 31;
  const int w    = tid >> 5;
  const int wm   = w >> 2;          // 0..1
  const int wn   = w & 3;           // 0..3
  const int mblk = blockIdx.y * 128;
  const int nblk = blockIdx.x * 128;
  const int m0   = mblk + wm * 64;
  const int n0   = nblk + wn * 32;

  v8f acc[4][2];
#pragma unroll
  for (int im = 0; im < 4; ++im)
#pragma unroll
    for (int jn = 0; jn < 2; ++jn) acc[im][jn] = vzero8f();

  if (CONV) {
    for (int kk = 0; kk < K; kk += 32) {
      const int kp = kk >> 10;          // conv tap 0..2
      const int i0 = kk & (DIM - 1);
      v16bf af[4], bfr[2];
#pragma unroll
      for (int im = 0; im < 4; ++im) {
        const int m = m0 + im * 16 + (lane & 15);
        const int s = (m & (SEQ - 1)) + kp - 1;   // shifted seq pos
        if (s >= 0 && s < SEQ) {
          const bf16* p = A + (size_t)(m + kp - 1) * DIM + i0 + ((lane >> 4) << 3);
          af[im] = cat8(*(const v8bf*)p, *(const v8bf*)(p + 16));
        } else {
          af[im] = vzero16bf();
        }
      }
#pragma unroll
      for (int jn = 0; jn < 2; ++jn)
        bfr[jn] = load_b_frag(Bt + (size_t)(n0 + jn * 16) * ldb + kk, ldb);
#pragma unroll
      for (int im = 0; im < 4; ++im)
#pragma unroll
        for (int jn = 0; jn < 2; ++jn)
          acc[im][jn] = wmma_bf16(af[im], bfr[jn], acc[im][jn]);
    }
  } else {
    __shared__ __align__(16) bf16 Asm[2][128 * LDSPAD];
    __shared__ __align__(16) bf16 Bsm[2][128 * LDSPAD];

    // Stage one 128x32 A tile and one 128x32 B tile into LDS buffer `buf`.
    // Each thread issues 2+2 b128 async copies -> 4 async instrs per wave.
    auto stage = [&](int buf, int kk) {
#pragma unroll
      for (int j = 0; j < 2; ++j) {
        const int c   = tid + j * 256;        // 512 chunks of 8 elements
        const int row = c >> 2;
        const int col = (c & 3) << 3;
        async_copy_b128(A  + (size_t)(mblk + row) * lda + kk + col,
                        &Asm[buf][row * LDSPAD + col]);
        async_copy_b128(Bt + (size_t)(nblk + row) * ldb + kk + col,
                        &Bsm[buf][row * LDSPAD + col]);
      }
    };

    const int nk = K >> 5;
    stage(0, 0);
    for (int kt = 0; kt < nk; ++kt) {
      const int cur = kt & 1;
      const bool more = (kt + 1) < nk;
      if (more) stage(cur ^ 1, (kt + 1) << 5);
      if (more) wait_async_le4(); else wait_async_zero();
      __syncthreads();

      v16bf af[4], bfr[2];
#pragma unroll
      for (int im = 0; im < 4; ++im)
        af[im] = load_a_frag(&Asm[cur][(wm * 64 + im * 16) * LDSPAD], LDSPAD);
#pragma unroll
      for (int jn = 0; jn < 2; ++jn)
        bfr[jn] = load_b_frag(&Bsm[cur][(wn * 32 + jn * 16) * LDSPAD], LDSPAD);
#pragma unroll
      for (int im = 0; im < 4; ++im)
#pragma unroll
        for (int jn = 0; jn < 2; ++jn)
          acc[im][jn] = wmma_bf16(af[im], bfr[jn], acc[im][jn]);

      __syncthreads();   // protect `cur` buffer from next iteration's staging
    }
  }

  // Epilogue. C layout: element i of lane L -> row (L>>4)*8 + i, col L&15.
#pragma unroll
  for (int im = 0; im < 4; ++im) {
#pragma unroll
    for (int jn = 0; jn < 2; ++jn) {
      const int col  = n0 + jn * 16 + (lane & 15);
      const int rowb = m0 + im * 16 + ((lane >> 4) << 3);
      const float bv = bias ? bias[col] : 0.0f;
#pragma unroll
      for (int i = 0; i < 8; ++i) {
        float v = acc[im][jn][i] + bv;
        if (relu) v = fmaxf(v, 0.0f);
        const size_t idx = (size_t)(rowb + i) * ldc + col;
        if (rowscale) {
          Cacc[idx] += rowscale[(size_t)(rowb + i) * rs_stride] * v;
        } else {
          if (Cf) Cf[idx] = v;
          if (Cb) Cb[idx] = (bf16)v;
        }
      }
    }
  }
}

// ---------------------------------------------------------------------------
// Flash attention: one wave (32 thr) per (bh, 16 q-rows). dh = 64.
// ---------------------------------------------------------------------------
__global__ __launch_bounds__(32) void attention_kernel(
    const bf16* __restrict__ qkv,   // (TOK, 3*DIM) bf16
    const bf16* __restrict__ vT,    // (B*H, DH, SEQ) bf16
    bf16* __restrict__ ctx)         // (TOK, DIM) bf16
{
  __shared__ __align__(16) bf16 psh[16 * 32];
  const int lane = threadIdx.x & 31;
  const int bh = blockIdx.y;
  const int b  = bh >> 4;
  const int h  = bh & 15;
  const int tb = b * SEQ;
  const int q0 = blockIdx.x * 16;
  const float scale = 0.125f;      // 1/sqrt(64)

  const bf16* qbase = qkv + (size_t)(tb + q0) * QKVLD + h * DH;
  const v16bf qa0 = load_a_frag(qbase, QKVLD);        // d 0..31
  const v16bf qa1 = load_a_frag(qbase + 32, QKVLD);   // d 32..63

  v8f accv[4];
#pragma unroll
  for (int j = 0; j < 4; ++j) accv[j] = vzero8f();
  float mrow[8], lrow[8];
#pragma unroll
  for (int i = 0; i < 8; ++i) { mrow[i] = -1e30f; lrow[i] = 0.0f; }

  const bf16* kbase = qkv + (size_t)tb * QKVLD + DIM + h * DH;
  const bf16* vbase = vT + (size_t)(bh * DH) * SEQ;
  const int prow = (lane >> 4) << 3;
  const int pcol = lane & 15;

  for (int k0 = 0; k0 < SEQ; k0 += 32) {
    // scores: 16 x 32, as two 16x16 C tiles (K^T fragments are B frags)
    v16bf kb00 = load_b_frag(kbase + (size_t)k0 * QKVLD, QKVLD);
    v16bf kb01 = load_b_frag(kbase + (size_t)k0 * QKVLD + 32, QKVLD);
    v16bf kb10 = load_b_frag(kbase + (size_t)(k0 + 16) * QKVLD, QKVLD);
    v16bf kb11 = load_b_frag(kbase + (size_t)(k0 + 16) * QKVLD + 32, QKVLD);
    v8f s0 = vzero8f(); s0 = wmma_bf16(qa0, kb00, s0); s0 = wmma_bf16(qa1, kb01, s0);
    v8f s1 = vzero8f(); s1 = wmma_bf16(qa0, kb10, s1); s1 = wmma_bf16(qa1, kb11, s1);

    // online softmax update (row = prow+i; 16 lanes of a half = 16 cols)
#pragma unroll
    for (int i = 0; i < 8; ++i) {
      float e0 = s0[i] * scale, e1 = s1[i] * scale;
      float mx = fmaxf(e0, e1);
#pragma unroll
      for (int off = 1; off < 16; off <<= 1) mx = fmaxf(mx, __shfl_xor(mx, off, 32));
      const float mnew = fmaxf(mrow[i], mx);
      const float p0 = __expf(e0 - mnew);
      const float p1 = __expf(e1 - mnew);
      const float corr = __expf(mrow[i] - mnew);
      float ps = p0 + p1;
#pragma unroll
      for (int off = 1; off < 16; off <<= 1) ps += __shfl_xor(ps, off, 32);
      lrow[i] = lrow[i] * corr + ps;
      mrow[i] = mnew;
      accv[0][i] *= corr; accv[1][i] *= corr; accv[2][i] *= corr; accv[3][i] *= corr;
      psh[(prow + i) * 32 + pcol]      = (bf16)p0;
      psh[(prow + i) * 32 + pcol + 16] = (bf16)p1;
    }
    __syncthreads();
    const v16bf pa = load_a_frag(psh, 32);   // C-layout -> A-layout via LDS
#pragma unroll
    for (int j = 0; j < 4; ++j) {
      v16bf vb = load_b_frag(vbase + (size_t)(j * 16) * SEQ + k0, SEQ);
      accv[j] = wmma_bf16(pa, vb, accv[j]);
    }
    __syncthreads();
  }

#pragma unroll
  for (int i = 0; i < 8; ++i) {
    const float inv = 1.0f / lrow[i];
    const int t = tb + q0 + prow + i;
#pragma unroll
    for (int j = 0; j < 4; ++j)
      ctx[(size_t)t * DIM + h * DH + j * 16 + pcol] = (bf16)(accv[j][i] * inv);
  }
}

// ---------------------------------------------------------------------------
// residual add + LayerNorm (one block per token)
// ---------------------------------------------------------------------------
__global__ __launch_bounds__(256) void add_ln_kernel(
    const float* __restrict__ a, const float* __restrict__ r,
    const float* __restrict__ g, const float* __restrict__ bta,
    float* __restrict__ outf, bf16* __restrict__ outb)
{
  __shared__ float xrow[DIM];
  __shared__ float red[256];
  const int t = blockIdx.x, tid = threadIdx.x;
  float s1 = 0.0f, s2 = 0.0f;
  for (int d = tid; d < DIM; d += 256) {
    const float v = a[(size_t)t * DIM + d] + r[(size_t)t * DIM + d];
    xrow[d] = v; s1 += v; s2 += v * v;
  }
  red[tid] = s1; __syncthreads();
  for (int st = 128; st > 0; st >>= 1) { if (tid < st) red[tid] += red[tid + st]; __syncthreads(); }
  const float mean = red[0] * (1.0f / DIM); __syncthreads();
  red[tid] = s2; __syncthreads();
  for (int st = 128; st > 0; st >>= 1) { if (tid < st) red[tid] += red[tid + st]; __syncthreads(); }
  const float var = red[0] * (1.0f / DIM) - mean * mean;
  const float rstd = rsqrtf(var + 1e-5f);
  for (int d = tid; d < DIM; d += 256) {
    const float y = (xrow[d] - mean) * rstd * g[d] + bta[d];
    if (outf) outf[(size_t)t * DIM + d] = y;
    if (outb) outb[(size_t)t * DIM + d] = (bf16)y;
  }
}

// ---------------------------------------------------------------------------
// MoE gate: probs = softmax(x @ gate_w + gate_b), one block per token
// ---------------------------------------------------------------------------
__global__ __launch_bounds__(256) void gate_kernel(
    const float* __restrict__ x, const float* __restrict__ gw,
    const float* __restrict__ gb, float* __restrict__ probs)
{
  __shared__ float red[256];
  __shared__ float ge[NEXP];
  const int t = blockIdx.x, tid = threadIdx.x;
  float acc[NEXP];
#pragma unroll
  for (int e = 0; e < NEXP; ++e) acc[e] = 0.0f;
  for (int d = tid; d < DIM; d += 256) {
    const float xv = x[(size_t)t * DIM + d];
#pragma unroll
    for (int e = 0; e < NEXP; ++e) acc[e] += xv * gw[d * NEXP + e];
  }
  for (int e = 0; e < NEXP; ++e) {
    red[tid] = acc[e]; __syncthreads();
    for (int st = 128; st > 0; st >>= 1) { if (tid < st) red[tid] += red[tid + st]; __syncthreads(); }
    if (tid == 0) ge[e] = red[0] + gb[e];
    __syncthreads();
  }
  if (tid == 0) {
    float mx = -1e30f;
#pragma unroll
    for (int e = 0; e < NEXP; ++e) mx = fmaxf(mx, ge[e]);
    float s = 0.0f; float pe[NEXP];
#pragma unroll
    for (int e = 0; e < NEXP; ++e) { pe[e] = __expf(ge[e] - mx); s += pe[e]; }
    const float inv = 1.0f / s;
#pragma unroll
    for (int e = 0; e < NEXP; ++e) probs[(size_t)t * NEXP + e] = pe[e] * inv;
  }
}

// ---------------------------------------------------------------------------
// elementwise helpers
// ---------------------------------------------------------------------------
__global__ void cvt_f32_bf16(const float* __restrict__ s, bf16* __restrict__ d, int n) {
  const int i = blockIdx.x * 256 + threadIdx.x;
  if (i < n) d[i] = (bf16)s[i];
}
__global__ void zero_f32(float* __restrict__ d, int n) {
  const int i = blockIdx.x * 256 + threadIdx.x;
  if (i < n) d[i] = 0.0f;
}
// conv_w (O,I,3) f32 -> cwT[o][k*DIM + i] bf16
__global__ void pack_convw(const float* __restrict__ w, bf16* __restrict__ dst) {
  const int idx = blockIdx.x * 256 + threadIdx.x;
  if (idx >= DIM * 3 * DIM) return;
  const int o = idx / (3 * DIM);
  const int r = idx - o * (3 * DIM);
  const int k = r >> 10;
  const int i = r & (DIM - 1);
  dst[idx] = (bf16)w[(size_t)(o * DIM + i) * 3 + k];
}
// qkv bf16 -> vT[(b*H+h)*DH + d][s]
__global__ void pack_vT(const bf16* __restrict__ qkv, bf16* __restrict__ vT) {
  const int idx = blockIdx.x * 256 + threadIdx.x;
  if (idx >= TOK * DIM) return;
  const int s = idx & (SEQ - 1);
  const int d = (idx >> 11) & (DH - 1);
  const int h = (idx >> 17) & (HEADS - 1);
  const int b = idx >> 21;
  vT[idx] = qkv[(size_t)(b * SEQ + s) * QKVLD + 2 * DIM + h * DH + d];
}
// w1[e] (DIM x DFF) f32 -> dst[f*DIM + d] bf16 (transpose)
__global__ void tcvt_w1(const float* __restrict__ w, bf16* __restrict__ dst) {
  const int idx = blockIdx.x * 256 + threadIdx.x;
  if (idx >= DFF * DIM) return;
  const int f = idx >> 10;
  const int d = idx & (DIM - 1);
  dst[idx] = (bf16)w[(size_t)d * DFF + f];
}
// w2[e] (DFF x DIM) f32 -> dst[d*DFF + f] bf16 (transpose)
__global__ void tcvt_w2(const float* __restrict__ w, bf16* __restrict__ dst) {
  const int idx = blockIdx.x * 256 + threadIdx.x;
  if (idx >= DIM * DFF) return;
  const int d = idx >> 12;
  const int f = idx & (DFF - 1);
  dst[idx] = (bf16)w[(size_t)f * DIM + d];
}

// ---------------------------------------------------------------------------
extern "C" void kernel_launch(void* const* d_in, const int* in_sizes, int n_in,
                              void* d_out, int out_size, void* d_ws, size_t ws_size,
                              hipStream_t stream)
{
  const float* x      = (const float*)d_in[0];
  const float* conv_w = (const float*)d_in[1];
  const float* conv_b = (const float*)d_in[2];
  const float* wqkv   = (const float*)d_in[3];
  const float* bqkv   = (const float*)d_in[4];
  const float* wo     = (const float*)d_in[5];
  const float* bo     = (const float*)d_in[6];
  const float* ln1_g  = (const float*)d_in[7];
  const float* ln1_b  = (const float*)d_in[8];
  const float* gate_w = (const float*)d_in[9];
  const float* gate_b = (const float*)d_in[10];
  const float* w1     = (const float*)d_in[11];
  const float* b1     = (const float*)d_in[12];
  const float* w2     = (const float*)d_in[13];
  const float* b2     = (const float*)d_in[14];
  const float* ln2_g  = (const float*)d_in[15];
  const float* ln2_b  = (const float*)d_in[16];

  char* ws = (char*)d_ws;
  size_t off = 0;
  auto carve = [&](size_t bytes) -> char* {
    char* p = ws + off;
    off = (off + bytes + 255) & ~(size_t)255;
    return p;
  };
  const size_t TD = (size_t)TOK * DIM;
  bf16*  xbf    = (bf16*)carve(TD * 2);
  float* xcf    = (float*)carve(TD * 4);
  bf16*  xcb    = (bf16*)carve(TD * 2);
  bf16*  qkvb   = (bf16*)carve(3 * TD * 2);
  bf16*  vT     = (bf16*)carve(TD * 2);
  bf16*  ctxb   = (bf16*)carve(TD * 2);
  float* aout   = (float*)carve(TD * 4);
  float* x1f    = (float*)carve(TD * 4);
  bf16*  x1b    = (bf16*)carve(TD * 2);
  float* probs  = (float*)carve((size_t)TOK * NEXP * 4);
  bf16*  hbuf   = (bf16*)carve((size_t)TOK * DFF * 2);
  float* moef   = (float*)carve(TD * 4);
  bf16*  wstage = (bf16*)carve((size_t)DFF * DIM * 2);
  (void)ws_size; (void)in_sizes; (void)n_in; (void)out_size;

  const int EB = 256;
  const int nTD = TOK * DIM;

  // 1) x -> bf16
  cvt_f32_bf16<<<(nTD + EB - 1) / EB, EB, 0, stream>>>(x, xbf, nTD);
  // 2) conv weights -> cwT (in wstage)
  pack_convw<<<(DIM * 3 * DIM + EB - 1) / EB, EB, 0, stream>>>(conv_w, wstage);
  // 3) conv as NT-GEMM (M=TOK, N=DIM, K=3*DIM), f32 + bf16 outputs
  gemm_nt_kernel<1><<<dim3(DIM / 128, TOK / 128), 256, 0, stream>>>(
      xbf, DIM, wstage, 3 * DIM, TOK, DIM, 3 * DIM,
      conv_b, xcf, xcb, DIM, 0, nullptr, 0, nullptr);
  // 4) wqkv -> bf16 (already "B^T": rows are output features over K=DIM)
  cvt_f32_bf16<<<(3 * DIM * DIM + EB - 1) / EB, EB, 0, stream>>>(wqkv, wstage, 3 * DIM * DIM);
  // 5) qkv = xc @ wqkv^T + bqkv  -> bf16
  gemm_nt_kernel<0><<<dim3(3 * DIM / 128, TOK / 128), 256, 0, stream>>>(
      xcb, DIM, wstage, DIM, TOK, 3 * DIM, DIM,
      bqkv, nullptr, qkvb, 3 * DIM, 0, nullptr, 0, nullptr);
  // 6) V -> vT (per (b,h): DH x SEQ)
  pack_vT<<<(nTD + EB - 1) / EB, EB, 0, stream>>>(qkvb, vT);
  // 7) flash attention -> ctxb
  attention_kernel<<<dim3(SEQ / 16, BATCH * HEADS), 32, 0, stream>>>(qkvb, vT, ctxb);
  // 8) wo -> bf16 ; attn_out = ctx @ wo^T + bo
  cvt_f32_bf16<<<(DIM * DIM + EB - 1) / EB, EB, 0, stream>>>(wo, wstage, DIM * DIM);
  gemm_nt_kernel<0><<<dim3(DIM / 128, TOK / 128), 256, 0, stream>>>(
      ctxb, DIM, wstage, DIM, TOK, DIM, DIM,
      bo, aout, nullptr, DIM, 0, nullptr, 0, nullptr);
  // 9) x1 = LN(xc + attn_out)
  add_ln_kernel<<<TOK, 256, 0, stream>>>(xcf, aout, ln1_g, ln1_b, x1f, x1b);
  // 10) gate probs
  gate_kernel<<<TOK, 256, 0, stream>>>(x1f, gate_w, gate_b, probs);
  // 11) moe accumulator = 0
  zero_f32<<<(nTD + EB - 1) / EB, EB, 0, stream>>>(moef, nTD);
  // 12) dense MoE: for each expert, h = relu(x1@w1[e]+b1[e]); moe += p_e*(h@w2[e]+b2[e])
  for (int e = 0; e < NEXP; ++e) {
    tcvt_w1<<<(DFF * DIM + EB - 1) / EB, EB, 0, stream>>>(w1 + (size_t)e * DIM * DFF, wstage);
    gemm_nt_kernel<0><<<dim3(DFF / 128, TOK / 128), 256, 0, stream>>>(
        x1b, DIM, wstage, DIM, TOK, DFF, DIM,
        b1 + (size_t)e * DFF, nullptr, hbuf, DFF, 1, nullptr, 0, nullptr);
    tcvt_w2<<<(DIM * DFF + EB - 1) / EB, EB, 0, stream>>>(w2 + (size_t)e * DFF * DIM, wstage);
    gemm_nt_kernel<0><<<dim3(DIM / 128, TOK / 128), 256, 0, stream>>>(
        hbuf, DFF, wstage, DFF, TOK, DIM, DFF,
        b2 + (size_t)e * DIM, nullptr, nullptr, DIM, 0, probs + e, NEXP, moef);
  }
  // 13) out = LN(x1 + moe)
  add_ln_kernel<<<TOK, 256, 0, stream>>>(x1f, moef, ln2_g, ln2_b, (float*)d_out, nullptr);
}
